// DCRNN_RGCN_89008902243183
// MI455X (gfx1250) — matrix operand order repaired and checked by reading, not compiled, not run on hardware
//
#include <hip/hip_runtime.h>
#include <hip/hip_bf16.h>
#include <math.h>

#define NNODE 50000
#define NEDGE 800000
#define FDIM  128
#define NCLS  16
#define NTILE (NNODE / 16)   // 3125, exact

typedef _Float16 f16;
typedef __attribute__((ext_vector_type(16))) _Float16 v16h;
typedef __attribute__((ext_vector_type(8)))  float    v8f;

// ---------------------------------------------------------------------------
// init: agg = 0, deg = 1 (self-loop weight)
__global__ void k_init(float* __restrict__ agg, float* __restrict__ deg) {
    int i = blockIdx.x * blockDim.x + threadIdx.x;
    if (i < NNODE * FDIM) agg[i] = 0.0f;
    if (i < NNODE)        deg[i] = 1.0f;
}

// deg[dst] += ew
__global__ void k_deg(const int* __restrict__ dst, const float* __restrict__ ew,
                      float* __restrict__ deg) {
    int e = blockIdx.x * blockDim.x + threadIdx.x;
    if (e < NEDGE) atomicAdd(&deg[dst[e]], ew[e]);
}

__global__ void k_dinv(const float* __restrict__ deg, float* __restrict__ dinv) {
    int i = blockIdx.x * blockDim.x + threadIdx.x;
    if (i < NNODE) {
        float d = deg[i];
        dinv[i] = (d > 0.0f) ? rsqrtf(d) : 0.0f;
    }
}

// weight prep: sum gate weight pairs, convert to f16, transpose to [n][k]
__global__ void k_wprep(const float* __restrict__ Wg,
                        const float* __restrict__ Wz0, const float* __restrict__ Wz1,
                        const float* __restrict__ Wr0, const float* __restrict__ Wr1,
                        const float* __restrict__ Wh0, const float* __restrict__ Wh1,
                        f16* __restrict__ WgT,
                        f16* __restrict__ WzAT, f16* __restrict__ WzBT,
                        f16* __restrict__ WrAT, f16* __restrict__ WrBT,
                        f16* __restrict__ WhAT, f16* __restrict__ WhBT) {
    int i = blockIdx.x * blockDim.x + threadIdx.x;
    if (i >= FDIM * FDIM) return;
    int k = i / FDIM, n = i % FDIM;
    int t  = n * FDIM + k;              // transposed index
    int lo = k * FDIM + n;              // rows 0..127 of the 256-row gate weights
    int hi = (k + FDIM) * FDIM + n;     // rows 128..255
    WgT[t]  = (f16)Wg[lo];
    WzAT[t] = (f16)(Wz0[lo] + Wz1[lo]);
    WzBT[t] = (f16)(Wz0[hi] + Wz1[hi]);
    WrAT[t] = (f16)(Wr0[lo] + Wr1[lo]);
    WrBT[t] = (f16)(Wr0[hi] + Wr1[hi]);
    WhAT[t] = (f16)(Wh0[lo] + Wh1[lo]);
    WhBT[t] = (f16)(Wh0[hi] + Wh1[hi]);
}

// ---------------------------------------------------------------------------
// A-fragment loader: 16x32 f16 A layout (wave32).
// lane = 16*half_ + r holds row (rowbase+r); elems 0..15 -> K = kbase+16*half_+i.
__device__ __forceinline__ v16h afrag(const float* __restrict__ rowp, int koff) {
    const float4* p = (const float4*)(rowp + koff);
    const float4* q = (const float4*)(rowp + koff + 8);
    float4 u0 = p[0], u1 = p[1], w0 = q[0], w1 = q[1];
    v16h a;
    a[0]=(f16)u0.x;  a[1]=(f16)u0.y;  a[2]=(f16)u0.z;  a[3]=(f16)u0.w;
    a[4]=(f16)u1.x;  a[5]=(f16)u1.y;  a[6]=(f16)u1.z;  a[7]=(f16)u1.w;
    a[8]=(f16)w0.x;  a[9]=(f16)w0.y;  a[10]=(f16)w0.z; a[11]=(f16)w0.w;
    a[12]=(f16)w1.x; a[13]=(f16)w1.y; a[14]=(f16)w1.z; a[15]=(f16)w1.w;
    return a;
}
__device__ __forceinline__ v16h afrag_mul(const float* __restrict__ rowp,
                                          const float* __restrict__ mulp, int koff) {
    const float4* p  = (const float4*)(rowp + koff);
    const float4* q  = (const float4*)(rowp + koff + 8);
    const float4* mp = (const float4*)(mulp + koff);
    const float4* mq = (const float4*)(mulp + koff + 8);
    float4 u0 = p[0], u1 = p[1], w0 = q[0], w1 = q[1];
    float4 m0 = mp[0], m1 = mp[1], n0 = mq[0], n1 = mq[1];
    v16h a;
    a[0]=(f16)(u0.x*m0.x);  a[1]=(f16)(u0.y*m0.y);  a[2]=(f16)(u0.z*m0.z);  a[3]=(f16)(u0.w*m0.w);
    a[4]=(f16)(u1.x*m1.x);  a[5]=(f16)(u1.y*m1.y);  a[6]=(f16)(u1.z*m1.z);  a[7]=(f16)(u1.w*m1.w);
    a[8]=(f16)(w0.x*n0.x);  a[9]=(f16)(w0.y*n0.y);  a[10]=(f16)(w0.z*n0.z); a[11]=(f16)(w0.w*n0.w);
    a[12]=(f16)(w1.x*n1.x); a[13]=(f16)(w1.y*n1.y); a[14]=(f16)(w1.z*n1.z); a[15]=(f16)(w1.w*n1.w);
    return a;
}

// Out[N,128] = act( A1@W1 + (A2 .* A2mul)@W2 + bias ), W*T pre-transposed f16 [n][k].
// Weights staged in LDS once per block (8 waves share); wave computes a 16x128
// strip: 4 K-steps x 8 N-tiles of v_wmma_f32_16x16x32_f16, straight-line
// (template-specialized: no branches inside the WMMA chain).
template <int HAS_A2, int HAS_MUL, int ACT>
__global__ void __launch_bounds__(256)
k_gemm(const float* __restrict__ A1, const f16* __restrict__ W1T,
       const float* __restrict__ A2, const f16* __restrict__ W2T,
       const float* __restrict__ A2mul,
       const float* __restrict__ bias, float* __restrict__ Out) {
    __shared__ f16 lds_w1[FDIM * FDIM];
    __shared__ f16 lds_w2[HAS_A2 ? FDIM * FDIM : 16];

    // cooperative weight staging: 32 KB per matrix, 128 B (8x b128) per thread
    {
        const uint4* s1 = (const uint4*)W1T;
        uint4* d1 = (uint4*)lds_w1;
        #pragma unroll
        for (int it = 0; it < 8; ++it) {
            int idx = it * 256 + threadIdx.x;   // 2048 uint4 = 32 KB
            d1[idx] = s1[idx];
        }
        if (HAS_A2) {
            const uint4* s2 = (const uint4*)W2T;
            uint4* d2 = (uint4*)lds_w2;
            #pragma unroll
            for (int it = 0; it < 8; ++it) {
                int idx = it * 256 + threadIdx.x;
                d2[idx] = s2[idx];
            }
        }
    }
    __syncthreads();

    int tile = blockIdx.x * 8 + (threadIdx.x >> 5);   // uniform per wave
    if (tile >= NTILE) return;                        // whole-wave exit: EXEC all-1s
    int lane  = threadIdx.x & 31;
    int half_ = lane >> 4;
    int r     = lane & 15;
    int rowbase = tile * 16;

    v8f acc[8];
    #pragma unroll
    for (int nt = 0; nt < 8; ++nt) {
        float b = bias ? bias[nt * 16 + r] : 0.0f;    // C/D col = lane%16
        #pragma unroll
        for (int j = 0; j < 8; ++j) acc[nt][j] = b;
    }

    const float* a1row = A1 + (size_t)(rowbase + r) * FDIM;
    const float* a2row = HAS_A2  ? A2    + (size_t)(rowbase + r) * FDIM : nullptr;
    const float* mrow  = HAS_MUL ? A2mul + (size_t)(rowbase + r) * FDIM : nullptr;

    #pragma unroll
    for (int kstep = 0; kstep < 4; ++kstep) {
        int kbase = kstep * 32;
        int koff  = kbase + 16 * half_;
        v16h a1 = afrag(a1row, koff);
        v16h a2;
        if (HAS_A2) a2 = HAS_MUL ? afrag_mul(a2row, mrow, koff) : afrag(a2row, koff);
        #pragma unroll
        for (int nt = 0; nt < 8; ++nt) {
            // B 32x16 layout: lane col = nt*16+r, K = kbase + 16*half_ + i (contiguous)
            v16h b1 = *(const v16h*)(lds_w1 + (nt * 16 + r) * FDIM + kbase + 16 * half_);
            acc[nt] = __builtin_amdgcn_wmma_f32_16x16x32_f16(
                false, a1, false, b1, (short)0, acc[nt], false, false);
            if (HAS_A2) {
                v16h b2 = *(const v16h*)(lds_w2 + (nt * 16 + r) * FDIM + kbase + 16 * half_);
                acc[nt] = __builtin_amdgcn_wmma_f32_16x16x32_f16(
                    false, a2, false, b2, (short)0, acc[nt], false, false);
            }
        }
    }

    #pragma unroll
    for (int nt = 0; nt < 8; ++nt) {
        #pragma unroll
        for (int j = 0; j < 8; ++j) {
            int row = rowbase + j + 8 * half_;
            float v = acc[nt][j];
            if (ACT == 1)      v = 1.0f / (1.0f + __expf(-v));
            else if (ACT == 2) v = tanhf(v);
            Out[(size_t)row * FDIM + nt * 16 + r] = v;
        }
    }
}

// ---------------------------------------------------------------------------
// edge scatter: wave per edge, lane handles 4 floats; L2-resident atomics
__global__ void k_scatter(const int* __restrict__ src, const int* __restrict__ dst,
                          const float* __restrict__ ew, const float* __restrict__ dinv,
                          const float* __restrict__ xw, float* __restrict__ agg) {
    int t = blockIdx.x * blockDim.x + threadIdx.x;
    int e = t >> 5;
    int lane = t & 31;
    if (e >= NEDGE) return;
    int s = src[e], d = dst[e];
    float coef = dinv[s] * ew[e] * dinv[d];
    float4 v = ((const float4*)(xw + (size_t)s * FDIM))[lane];
    float* a = agg + (size_t)d * FDIM + lane * 4;
    atomicAdd(a + 0, coef * v.x);
    atomicAdd(a + 1, coef * v.y);
    atomicAdd(a + 2, coef * v.z);
    atomicAdd(a + 3, coef * v.w);
}

// h = agg + dinv^2 * xw (self loop) + bg
__global__ void k_hfin(float* __restrict__ h, const float* __restrict__ xw,
                       const float* __restrict__ dinv, const float* __restrict__ bg) {
    int i = blockIdx.x * blockDim.x + threadIdx.x;
    if (i >= NNODE * FDIM) return;
    int nn = i >> 7, f = i & 127;
    float di = dinv[nn];
    h[i] = h[i] + di * di * xw[i] + bg[f];
}

// Hn = Z*Hs + (1-Z)*Ht
__global__ void k_hn(const float* __restrict__ Z, const float* __restrict__ Hs,
                     const float* __restrict__ Ht, float* __restrict__ Hn) {
    int i = blockIdx.x * blockDim.x + threadIdx.x;
    if (i >= NNODE * FDIM) return;
    float z = Z[i];
    Hn[i] = z * Hs[i] + (1.0f - z) * Ht[i];
}

// probs = softmax(relu(Hn)@Wl + bl): 16 threads per node, shfl_xor width-16 softmax
__global__ void k_head(const float* __restrict__ Hn, const float* __restrict__ Wl,
                       const float* __restrict__ bl, float* __restrict__ probs) {
    int t = blockIdx.x * blockDim.x + threadIdx.x;
    int node = t >> 4;
    int c = t & 15;
    if (node >= NNODE) return;
    const float* hr = Hn + (size_t)node * FDIM;
    float acc = bl[c];
    #pragma unroll 4
    for (int f = 0; f < FDIM; ++f) {
        float v = hr[f];
        v = v > 0.0f ? v : 0.0f;
        acc += v * Wl[f * NCLS + c];
    }
    float m = acc;
    #pragma unroll
    for (int o = 8; o >= 1; o >>= 1) m = fmaxf(m, __shfl_xor(m, o, 16));
    float e = __expf(acc - m);
    float s = e;
    #pragma unroll
    for (int o = 8; o >= 1; o >>= 1) s += __shfl_xor(s, o, 16);
    probs[(size_t)node * NCLS + c] = e / s;
}

// ---------------------------------------------------------------------------
extern "C" void kernel_launch(void* const* d_in, const int* in_sizes, int n_in,
                              void* d_out, int out_size, void* d_ws, size_t ws_size,
                              hipStream_t stream) {
    const float* x   = (const float*)d_in[0];
    const int*   ei  = (const int*)  d_in[1];
    const float* ew  = (const float*)d_in[2];
    const float* Hs  = (const float*)d_in[3];
    const float* Wg  = (const float*)d_in[4];
    const float* bg  = (const float*)d_in[5];
    const float* Wz0 = (const float*)d_in[6];
    const float* Wz1 = (const float*)d_in[7];
    const float* bz  = (const float*)d_in[8];
    const float* Wr0 = (const float*)d_in[9];
    const float* Wr1 = (const float*)d_in[10];
    const float* br  = (const float*)d_in[11];
    const float* Wh0 = (const float*)d_in[12];
    const float* Wh1 = (const float*)d_in[13];
    const float* bh  = (const float*)d_in[14];
    const float* Wl  = (const float*)d_in[15];
    const float* bl  = (const float*)d_in[16];
    const int* src = ei;
    const int* dst = ei + NEDGE;

    unsigned char* w = (unsigned char*)d_ws;
    auto carve = [&](size_t bytes) {
        void* p = (void*)w;
        w += (bytes + 255) & ~(size_t)255;
        return p;
    };
    float* deg  = (float*)carve((size_t)NNODE * 4);
    float* dinv = (float*)carve((size_t)NNODE * 4);
    f16* WgT  = (f16*)carve((size_t)FDIM * FDIM * 2);
    f16* WzAT = (f16*)carve((size_t)FDIM * FDIM * 2);
    f16* WzBT = (f16*)carve((size_t)FDIM * FDIM * 2);
    f16* WrAT = (f16*)carve((size_t)FDIM * FDIM * 2);
    f16* WrBT = (f16*)carve((size_t)FDIM * FDIM * 2);
    f16* WhAT = (f16*)carve((size_t)FDIM * FDIM * 2);
    f16* WhBT = (f16*)carve((size_t)FDIM * FDIM * 2);
    float* xw  = (float*)carve((size_t)NNODE * FDIM * 4);
    float* agg = (float*)carve((size_t)NNODE * FDIM * 4);  // becomes h in place
    float* Zb  = (float*)carve((size_t)NNODE * FDIM * 4);
    float* Rb  = (float*)carve((size_t)NNODE * FDIM * 4);
    float* Htb = xw;                                       // xw dead after k_hfin

    float* outProbs = (float*)d_out;
    float* outHn    = (float*)d_out + (size_t)NNODE * NCLS;

    const int B = 256;
    int gNF = (NNODE * FDIM + B - 1) / B;

    k_init<<<gNF, B, 0, stream>>>(agg, deg);
    k_deg<<<(NEDGE + B - 1) / B, B, 0, stream>>>(dst, ew, deg);
    k_dinv<<<(NNODE + B - 1) / B, B, 0, stream>>>(deg, dinv);
    k_wprep<<<(FDIM * FDIM + B - 1) / B, B, 0, stream>>>(
        Wg, Wz0, Wz1, Wr0, Wr1, Wh0, Wh1, WgT, WzAT, WzBT, WrAT, WrBT, WhAT, WhBT);

    int gG = (NTILE + 7) / 8;
    // xw = x @ Wg
    k_gemm<0, 0, 0><<<gG, B, 0, stream>>>(x, WgT, nullptr, nullptr, nullptr, nullptr, xw);
    // GCN message passing
    k_scatter<<<(NEDGE * 32 + B - 1) / B, B, 0, stream>>>(src, dst, ew, dinv, xw, agg);
    k_hfin<<<gNF, B, 0, stream>>>(agg, xw, dinv, bg);
    // gates: Z = sigmoid(h@WzA + Hs@WzB + bz), R likewise
    k_gemm<1, 0, 1><<<gG, B, 0, stream>>>(agg, WzAT, Hs, WzBT, nullptr, bz, Zb);
    k_gemm<1, 0, 1><<<gG, B, 0, stream>>>(agg, WrAT, Hs, WrBT, nullptr, br, Rb);
    // Ht = tanh(h@WhA + (Hs.*R)@WhB + bh)   (reuses xw buffer)
    k_gemm<1, 1, 2><<<gG, B, 0, stream>>>(agg, WhAT, Hs, WhBT, Rb, bh, Htb);
    // Hn and head
    k_hn<<<gNF, B, 0, stream>>>(Zb, Hs, Htb, outHn);
    k_head<<<(NNODE * NCLS + B - 1) / B, B, 0, stream>>>(outHn, Wl, bl, outProbs);
}
